// PairTradingLSTM_62912680951970
// MI455X (gfx1250) — compile-verified, matrix operand order checked by reading, not run on hardware
//
#include <hip/hip_runtime.h>

// ---------------------------------------------------------------------------
// Persistent-workgroup fused 2-layer LSTM + autoregressive prediction LSTM.
//   - 16 workgroups x 512 threads (16 wave32s); each WG owns 16 batch rows
//     end-to-end (no grid sync needed). 4 waves/SIMD for L2-latency hiding.
//   - f16 WMMA (v_wmma_f32_16x16x32_f16) with fp32 accumulators; c-state in VGPRs.
//   - Weights pre-packed (pack_weights kernel) into WMMA B-fragment layout:
//     frag(kt,nt) is 1024B contiguous; lane reads its 32B with 2x global_load_b128.
//   - Per wave: 8 gate tiles (acc = 64 VGPRs), 8-fragment B staging; total
//     ~210 VGPRs/wave -> no s_set_vgpr_msb churn, deeper wave interleave.
//   - A (h-state) lives in LDS as f16, row stride 1032 halves (bank-conflict pad).
//   - Gate biases / K=1 input-weight columns hoisted to registers per phase.
// ---------------------------------------------------------------------------

typedef __attribute__((ext_vector_type(16))) _Float16 v16h;
typedef __attribute__((ext_vector_type(8)))  _Float16 v8h;
typedef __attribute__((ext_vector_type(8)))  float    v8f;

#define ASTRIDE 1032   // halves per A row (1024 + 8 pad -> 2064B, bank-spread)
#define HDIM    512
#define GDIM    2048   // 4*H gate width
#define TLEN    512
#define FUTURE  64
#define TPB     512    // threads per block
#define NW      16     // waves per block
#define SPW     2      // 16-col tiles per gate per wave (32 h-cols per wave)

__device__ __forceinline__ float sigm_(float x) { return 1.0f / (1.0f + __expf(-x)); }
__device__ __forceinline__ float tanh_(float x) {
    float e = __expf(2.0f * x);
    return 1.0f - 2.0f / (e + 1.0f);
}

__device__ __forceinline__ void zero_acc(v8f (&acc)[4][SPW]) {
#pragma unroll
    for (int q = 0; q < 4; q++)
#pragma unroll
        for (int s = 0; s < SPW; s++)
            acc[q][s] = v8f{0.f, 0.f, 0.f, 0.f, 0.f, 0.f, 0.f, 0.f};
}

// A-fragment & B-fragment half layout (consistent with pack_weights):
//   lanes 0-15 : M/N = lane, k_local = {0..7} (r0..3) and {16..23} (r4..7)
//   lanes16-31 : M/N = lane-16, k_local = {8..15} and {24..31}
__device__ __forceinline__ void gemm_acc(const _Float16* __restrict__ pack,
                                         const _Float16* A, int KT, int lane,
                                         int wv, v8f (&acc)[4][SPW]) {
    const int ln = lane & 15, hi = lane >> 4;
#pragma unroll 1
    for (int kt = 0; kt < KT; ++kt) {
        const _Float16* ar = A + ln * ASTRIDE + kt * 32 + hi * 8;
        v8h alo = *(const v8h*)(ar);
        v8h ahi = *(const v8h*)(ar + 16);
        v16h av = __builtin_shufflevector(alo, ahi, 0, 1, 2, 3, 4, 5, 6, 7, 8,
                                          9, 10, 11, 12, 13, 14, 15);
        const _Float16* bbase = pack + ((size_t)kt * 128) * 512 + lane * 16;
        // Stage the 8 B fragments of this kt before the WMMA chain.
        v16h bv[4 * SPW];
#pragma unroll
        for (int u = 0; u < 4 * SPW; ++u) {
            const int q = u / SPW, s = u % SPW;
            const _Float16* bp = bbase + (size_t)(q * 32 + wv * SPW + s) * 512;
            v8h blo = *(const v8h*)bp;
            v8h bhi = *(const v8h*)(bp + 8);
            bv[u] = __builtin_shufflevector(blo, bhi, 0, 1, 2, 3, 4, 5, 6, 7, 8,
                                            9, 10, 11, 12, 13, 14, 15);
        }
#pragma unroll
        for (int u = 0; u < 4 * SPW; ++u)
            acc[u / SPW][u % SPW] = __builtin_amdgcn_wmma_f32_16x16x32_f16(
                false, av, false, bv[u], (short)0, acc[u / SPW][u % SPW], false,
                false);
    }
}

// Hoist per-lane gate constants from LDS into registers once per phase.
__device__ __forceinline__ void load_gate_consts(const float* bias,
                                                 const float* wx, int lane,
                                                 int wv, float (&bg)[4][SPW],
                                                 float (&wg)[4][SPW]) {
    const int ln = lane & 15;
#pragma unroll
    for (int q = 0; q < 4; q++)
#pragma unroll
        for (int s = 0; s < SPW; s++) {
            const int jn = q * 512 + wv * (16 * SPW) + s * 16 + ln;
            bg[q][s] = bias[jn];
            wg[q][s] = wx[jn];
        }
}

__device__ __forceinline__ void load_bias_consts(const float* bias, int lane,
                                                 int wv, float (&bg)[4][SPW]) {
    const int ln = lane & 15;
#pragma unroll
    for (int q = 0; q < 4; q++)
#pragma unroll
        for (int s = 0; s < SPW; s++)
            bg[q][s] = bias[q * 512 + wv * (16 * SPW) + s * 16 + ln];
}

// Gate nonlinearity + c-state update (VGPR-resident) + h writeback to LDS (f16).
template <bool HAS_X>
__device__ __forceinline__ void gates_and_store(v8f (&acc)[4][SPW],
                                                const float (&bg)[4][SPW],
                                                const float (&wg)[4][SPW],
                                                const float* xp, int xstride,
                                                float (&cst)[SPW][8],
                                                _Float16* A, int hoff, int lane,
                                                int wv) {
    const int ln = lane & 15, hi = lane >> 4;
    float hv[SPW][8];
#pragma unroll
    for (int s = 0; s < SPW; s++) {
#pragma unroll
        for (int v = 0; v < 8; v++) {
            float xm = HAS_X ? xp[(v + hi * 8) * xstride] : 0.f;
            float gi = acc[0][s][v] + bg[0][s];
            float gf = acc[1][s][v] + bg[1][s];
            float gg = acc[2][s][v] + bg[2][s];
            float go = acc[3][s][v] + bg[3][s];
            if (HAS_X) {
                gi += xm * wg[0][s];
                gf += xm * wg[1][s];
                gg += xm * wg[2][s];
                go += xm * wg[3][s];
            }
            float cc = sigm_(gf) * cst[s][v] + sigm_(gi) * tanh_(gg);
            cst[s][v] = cc;
            hv[s][v] = sigm_(go) * tanh_(cc);
        }
    }
    __syncthreads();  // all waves done reading A for this GEMM
#pragma unroll
    for (int s = 0; s < SPW; s++) {
        const int jn = wv * (16 * SPW) + s * 16 + ln;
#pragma unroll
        for (int v = 0; v < 8; v++)
            A[(v + hi * 8) * ASTRIDE + hoff + jn] = (_Float16)hv[s][v];
    }
    __syncthreads();  // new h visible to all waves
}

// Output head: o[m] = h1[m,:] . w + b ; writes out[b0+m][sidx] and ovec[m].
__device__ __forceinline__ void head_out(const _Float16* A, const float* wlds,
                                         const float* bptr, float* red,
                                         float* ovec, float* out, int sidx,
                                         int b0, int tid) {
    const int m = tid >> 5, seg = tid & 31;  // 32 segments x 16 k each
    float p = 0.f;
    const int kb = seg * 16;
#pragma unroll 8
    for (int k = kb; k < kb + 16; k++)
        p += (float)A[m * ASTRIDE + 512 + k] * wlds[k];
    red[tid] = p;
    __syncthreads();
    if (seg == 0) {
        float sum = bptr[0];
        for (int i = 0; i < 32; i++) sum += red[m * 32 + i];
        ovec[m] = sum;
        out[(b0 + m) * FUTURE + sidx] = sum;
    }
    __syncthreads();
}

// Pack W (row-major [2048][K], optionally concat of two K0/K-K0 sources along K)
// into f16 WMMA B-fragments: dst[(kt*128+nt)*512 + lane*16 + r*2 + half].
__global__ void pack_weights(const float* __restrict__ srcA, int K0,
                             const float* __restrict__ srcB, int K,
                             _Float16* __restrict__ dst) {
    int e = blockIdx.x * 256 + threadIdx.x;
    if (e >= GDIM * K) return;
    int f = e >> 9;
    int r9 = e & 511;
    int lane = r9 >> 4;
    int idx = r9 & 15;
    int r = idx >> 1;
    int half = idx & 1;
    int kt = f >> 7;
    int nt = f & 127;
    int n = (nt << 4) + (lane & 15);
    int klocal = ((r >> 2) << 4) + ((lane >> 4) << 3) + ((r & 3) << 1) + half;
    int k = kt * 32 + klocal;
    float v = (k < K0) ? srcA[(size_t)n * K0 + k]
                       : srcB[(size_t)n * (K - K0) + (k - K0)];
    dst[e] = (_Float16)v;
}

__global__ __launch_bounds__(TPB, 1) void lstm_persistent(
    const float* __restrict__ x_g, const _Float16* __restrict__ packW0,
    const _Float16* __restrict__ packW1, const _Float16* __restrict__ packP0,
    const _Float16* __restrict__ packP1, const float* __restrict__ wih0,
    const float* __restrict__ bih0, const float* __restrict__ bhh0,
    const float* __restrict__ bih1, const float* __restrict__ bhh1,
    const float* __restrict__ fcw_g, const float* __restrict__ fcb_g,
    const float* __restrict__ pwih0, const float* __restrict__ pbih0,
    const float* __restrict__ pbhh0, const float* __restrict__ pbih1,
    const float* __restrict__ pbhh1, const float* __restrict__ pfcw_g,
    const float* __restrict__ pfcb_g, float* __restrict__ out) {
    extern __shared__ char smem[];
    _Float16* A  = (_Float16*)smem;              // 16*ASTRIDE f16  (33024 B)
    float* bias0 = (float*)(smem + 16 * ASTRIDE * 2);
    float* bias1 = bias0 + GDIM;
    float* pb0   = bias1 + GDIM;
    float* pb1   = pb0 + GDIM;
    float* w0v   = pb1 + GDIM;
    float* p0v   = w0v + GDIM;
    float* fcw   = p0v + GDIM;
    float* pfcw  = fcw + HDIM;
    float* xs    = pfcw + HDIM;                  // [16][512] f32
    float* red   = xs + 16 * TLEN;               // [TPB]
    float* ovec  = red + TPB;                    // [16]

    const int tid = threadIdx.x;
    const int lane = tid & 31;
    const int wv = tid >> 5;
    const int b0 = blockIdx.x * 16;

    for (int i = tid; i < GDIM; i += TPB) {
        bias0[i] = bih0[i] + bhh0[i];
        bias1[i] = bih1[i] + bhh1[i];
        pb0[i] = pbih0[i] + pbhh0[i];
        pb1[i] = pbih1[i] + pbhh1[i];
        w0v[i] = wih0[i];
        p0v[i] = pwih0[i];
    }
    for (int i = tid; i < HDIM; i += TPB) { fcw[i] = fcw_g[i]; pfcw[i] = pfcw_g[i]; }
    for (int i = tid; i < 16 * ASTRIDE; i += TPB) A[i] = (_Float16)0.f;
    for (int i = tid; i < 16 * TLEN; i += TPB) {
        int m = i >> 9, t = i & 511;
        xs[i] = x_g[((size_t)(b0 + m)) * TLEN + t];
    }
    float c0[SPW][8], c1[SPW][8];
#pragma unroll
    for (int s = 0; s < SPW; s++)
#pragma unroll
        for (int v = 0; v < 8; v++) { c0[s][v] = 0.f; c1[s][v] = 0.f; }
    __syncthreads();

    float bg0[4][SPW], wg0[4][SPW], bg1[4][SPW];
    load_gate_consts(bias0, w0v, lane, wv, bg0, wg0);
    load_bias_consts(bias1, lane, wv, bg1);

    v8f acc[4][SPW];

    // ---- warmup over observed sequence: fused layer0 -> layer1 per step ----
    for (int t = 0; t < TLEN; t++) {
        zero_acc(acc);
        gemm_acc(packW0, A, 16, lane, wv, acc);               // h0 @ Whh0^T
        gates_and_store<true>(acc, bg0, wg0, xs + t, TLEN, c0, A, 0, lane, wv);
        zero_acc(acc);
        gemm_acc(packW1, A, 32, lane, wv, acc);               // [h0,h1] @ [Wih1;Whh1]^T
        gates_and_store<false>(acc, bg1, wg0, nullptr, 0, c1, A, 512, lane, wv);
    }
    head_out(A, fcw, fcb_g, red, ovec, out, 0, b0, tid);      // out[:,0]

    // ---- switch gate constants to the prediction LSTM ----
    load_gate_consts(pb0, p0v, lane, wv, bg0, wg0);
    load_bias_consts(pb1, lane, wv, bg1);

    // ---- autoregressive decode: states/c-regs continue from warmup ----
    for (int s = 1; s < FUTURE; s++) {
        zero_acc(acc);
        gemm_acc(packP0, A, 16, lane, wv, acc);
        gates_and_store<true>(acc, bg0, wg0, ovec, 1, c0, A, 0, lane, wv);
        zero_acc(acc);
        gemm_acc(packP1, A, 32, lane, wv, acc);
        gates_and_store<false>(acc, bg1, wg0, nullptr, 0, c1, A, 512, lane, wv);
        head_out(A, pfcw, pfcb_g, red, ovec, out, s, b0, tid);
    }
}

extern "C" void kernel_launch(void* const* d_in, const int* in_sizes, int n_in,
                              void* d_out, int out_size, void* d_ws,
                              size_t ws_size, hipStream_t stream) {
    (void)in_sizes; (void)n_in; (void)out_size; (void)ws_size;
    const float* x      = (const float*)d_in[0];
    // d_in[1] = future (== 64, fixed by the reference)
    const float* w_ih0  = (const float*)d_in[2];
    const float* w_hh0  = (const float*)d_in[3];
    const float* b_ih0  = (const float*)d_in[4];
    const float* b_hh0  = (const float*)d_in[5];
    const float* w_ih1  = (const float*)d_in[6];
    const float* w_hh1  = (const float*)d_in[7];
    const float* b_ih1  = (const float*)d_in[8];
    const float* b_hh1  = (const float*)d_in[9];
    const float* fc_w   = (const float*)d_in[10];
    const float* fc_b   = (const float*)d_in[11];
    const float* pw_ih0 = (const float*)d_in[12];
    const float* pw_hh0 = (const float*)d_in[13];
    const float* pb_ih0 = (const float*)d_in[14];
    const float* pb_hh0 = (const float*)d_in[15];
    const float* pw_ih1 = (const float*)d_in[16];
    const float* pw_hh1 = (const float*)d_in[17];
    const float* pb_ih1 = (const float*)d_in[18];
    const float* pb_hh1 = (const float*)d_in[19];
    const float* pfc_w  = (const float*)d_in[20];
    const float* pfc_b  = (const float*)d_in[21];

    char* ws = (char*)d_ws;  // 12 MB of f16 fragment packs
    _Float16* packW0 = (_Float16*)(ws + 0);
    _Float16* packW1 = (_Float16*)(ws + (size_t)2 * 1024 * 1024);
    _Float16* packP0 = (_Float16*)(ws + (size_t)6 * 1024 * 1024);
    _Float16* packP1 = (_Float16*)(ws + (size_t)8 * 1024 * 1024);

    dim3 blk(256);
    pack_weights<<<(GDIM * 512) / 256, blk, 0, stream>>>(w_hh0, 512, w_hh0, 512, packW0);
    pack_weights<<<(GDIM * 1024) / 256, blk, 0, stream>>>(w_ih1, 512, w_hh1, 1024, packW1);
    pack_weights<<<(GDIM * 512) / 256, blk, 0, stream>>>(pw_hh0, 512, pw_hh0, 512, packP0);
    pack_weights<<<(GDIM * 1024) / 256, blk, 0, stream>>>(pw_ih1, 512, pw_hh1, 1024, packP1);

    // Dynamic LDS: A(33024) + 4 bias(32768) + 2 wvec(16384) + 2 fcw(4096)
    //            + xs(32768) + red(2048) + ovec(64) = 121152 B (<= 320KB/WGP)
    size_t shmem = 16 * ASTRIDE * 2 + 4 * GDIM * 4 + 2 * GDIM * 4 +
                   2 * HDIM * 4 + 16 * TLEN * 4 + TPB * 4 + 16 * 4;
    lstm_persistent<<<16, TPB, shmem, stream>>>(
        x, packW0, packW1, packP0, packP1, w_ih0, b_ih0, b_hh0, b_ih1, b_hh1,
        fc_w, fc_b, pw_ih0, pb_ih0, pb_hh0, pb_ih1, pb_hh1, pfc_w, pfc_b,
        (float*)d_out);
}
